// DiscriminatorBlock_38878043963811
// MI455X (gfx1250) — matrix operand order repaired and checked by reading, not compiled, not run on hardware
//
#include <hip/hip_runtime.h>
#include <hip/hip_bf16.h>
#include <math.h>

// ---------------- static config (mirrors reference) ----------------
#define IMG_B 16
#define IMG_C 3
#define IN_C  256
#define OUT_C 512
#define S     128
#define SOUT  64
#define HIDDEN 256
#define KGEN   32
#define NTAPS  28

// WMMA vector types (gfx1250)
typedef __attribute__((ext_vector_type(16))) __bf16 v16bf;
typedef __attribute__((ext_vector_type(8)))  float  v8f;

__device__ __forceinline__ unsigned short f2bf(float f) {
  union { float f; unsigned int u; } c; c.f = f;
  unsigned int u = c.u;
  return (unsigned short)((u + 0x7FFFu + ((u >> 16) & 1u)) >> 16); // RNE
}

// =====================================================================
// K1: generate the 3-tap separable depthwise filters (hz + vt).
// One workgroup, 256 threads (thread i owns output channel i).
// =====================================================================
__global__ __launch_bounds__(256) void gen_taps_kernel(
    const float* __restrict__ freqs,      // [256,2]
    const float* __restrict__ phases,     // [256]
    const float* __restrict__ hz_outdim,  // [256,256]
    const float* __restrict__ vt_outdim,  // [256,256]
    const float* __restrict__ gauss_sigma,// [1]
    const float* __restrict__ low_filter, // [28]
    const float* __restrict__ gauss_x,    // [32]
    float* __restrict__ taps_h,           // [256,3] out (pre-scaled by 1/sqrt(3))
    float* __restrict__ taps_v)           // [256,3] out
{
  __shared__ float wbuf[HIDDEN][KGEN];   // basis, then w
  __shared__ float red[256];
  __shared__ float colscale[KGEN];
  __shared__ float gwin[KGEN];
  __shared__ float lf[NTAPS];
  const int tid = threadIdx.x;

  float sigma = gauss_sigma[0];
  float gs = 1.0f + sigma * sigma * 0.2f;
  if (tid < NTAPS) lf[tid] = low_filter[tid];
  if (tid < KGEN)  gwin[tid] = expf(-(gauss_x[tid] * gauss_x[tid]) / (2.0f * gs));

  for (int d = 0; d < 2; ++d) {
    const float* coeff = d ? vt_outdim : hz_outdim;
    float* outp = d ? taps_v : taps_h;
    float fr = freqs[tid * 2 + d];
    float ph = phases[tid];
    __syncthreads();
    // basis[h][g] = sin((fr*g + ph)*2pi) * exp(-0.5)
    #pragma unroll
    for (int g = 0; g < KGEN; ++g) {
      float gv = ((float)g - 15.5f) * (2.0f / 33.0f);
      wbuf[tid][g] = sinf((fr * gv + ph) * 6.28318530718f) * 0.60653066f;
    }
    __syncthreads();
    // w[i][g] = sum_h basis[h][g]*coeff[i][h] / 16
    float wr[KGEN];
    #pragma unroll
    for (int g = 0; g < KGEN; ++g) wr[g] = 0.f;
    for (int h = 0; h < HIDDEN; ++h) {
      float cc = coeff[tid * HIDDEN + h];
      #pragma unroll
      for (int g = 0; g < KGEN; ++g) wr[g] += cc * wbuf[h][g];
    }
    float ssum = 0.f;
    #pragma unroll
    for (int g = 0; g < KGEN; ++g) { wr[g] *= 0.0625f; ssum += wr[g]; }
    // global mean over 256*32 elements
    red[tid] = ssum;
    __syncthreads();
    for (int off = 128; off > 0; off >>= 1) {
      if (tid < off) red[tid] += red[tid + off];
      __syncthreads();
    }
    float mean = red[0] * (1.0f / 8192.0f);
    __syncthreads();
    // subtract mean, stash w for per-column RMS
    #pragma unroll
    for (int g = 0; g < KGEN; ++g) { wr[g] -= mean; wbuf[tid][g] = wr[g]; }
    __syncthreads();
    if (tid < KGEN) {
      float s = 0.f;
      for (int i = 0; i < HIDDEN; ++i) { float v = wbuf[i][tid]; s += v * v; }
      colscale[tid] = rsqrtf(s * (1.0f / 256.0f) + 1e-8f);
    }
    __syncthreads();
    #pragma unroll
    for (int g = 0; g < KGEN; ++g) wr[g] *= colscale[g] * gwin[g];
    // valid correlation with 28-tap low-pass, take k = 0,2,4; fold in gdw=1/sqrt(3)
    #pragma unroll
    for (int k = 0; k < 3; ++k) {
      float a = 0.f;
      #pragma unroll
      for (int t = 0; t < NTAPS; ++t) a += wr[2 * k + t] * lf[t];
      outp[tid * 3 + k] = a * 0.57735026919f;
    }
  }
}

// =====================================================================
// K1b: Wcat[o][0:256]   = point_w[o][:] / 16            (consumes dw(x))
//      Wcat[o][256:512] = (point_w/16 . lr_w/16)[o][:]  (consumes x)
// =====================================================================
__global__ __launch_bounds__(256) void wcat_kernel(
    const float* __restrict__ point_w,   // [512,256]
    const float* __restrict__ lr_w,      // [256,256]
    unsigned short* __restrict__ Wcat)   // [512,512] bf16
{
  __shared__ float wp[IN_C];
  const int o = blockIdx.x;
  const int c = threadIdx.x;
  wp[c] = point_w[o * IN_C + c] * 0.0625f;
  __syncthreads();
  Wcat[o * 512 + c] = f2bf(wp[c]);
  float s = 0.f;
  for (int m = 0; m < IN_C; ++m) s += wp[m] * lr_w[m * IN_C + c];
  Wcat[o * 512 + 256 + c] = f2bf(s * 0.0625f);
}

// =====================================================================
// K2: fused  x = clamp(fromrgb(sin(img)))  and  y = dwH(dwV(x)).
// Block = (batch, 4-row strip); recomputes 1-px halo of x from img.
// Outputs are CHANNEL-LAST bf16: [b][y][x][c] so K4 can stage WMMA
// B-tiles with contiguous 64B chunks. 16-channel LDS chunking gives
// coalesced uint4 global stores.
// =====================================================================
__global__ __launch_bounds__(256) void fromrgb_dw_kernel(
    const float* __restrict__ img,       // [16,3,128,128]
    const float* __restrict__ rgb_w,     // [256,3]
    const float* __restrict__ rgb_b,     // [256]
    const float* __restrict__ taps_h,    // [256,3]
    const float* __restrict__ taps_v,    // [256,3]
    unsigned short* __restrict__ xout,   // [16,128,128,256] bf16 channel-last
    unsigned short* __restrict__ yout)   // [16,128,128,256] bf16 channel-last
{
  __shared__ float simg[3][6][132];      // 6 rows (halo), 130 used cols
  __shared__ float xt[6][132];
  __shared__ float wr[3 * IN_C];
  __shared__ float th[3 * IN_C];
  __shared__ float tv[3 * IN_C];
  __shared__ float bb[IN_C];
  __shared__ __align__(16) unsigned short xstage[512][16];  // 16 KB
  __shared__ __align__(16) unsigned short ystage[512][16];  // 16 KB
  const int tid = threadIdx.x;
  const int b = blockIdx.y;
  const int r0 = blockIdx.x * 4;

  for (int i = tid; i < 3 * IN_C; i += 256) {
    wr[i] = rgb_w[i] * 0.57735026919f;   // 1/sqrt(IMG_C)
    th[i] = taps_h[i];
    tv[i] = taps_v[i];
  }
  bb[tid] = rgb_b[tid];
  for (int i = tid; i < 3 * 6 * 132; i += 256) {
    int ch = i / 792, rem = i - ch * 792;
    int row = rem / 132, col = rem - row * 132;
    int gy = r0 - 1 + row, gx = col - 1;
    float v = 0.f;
    if (gy >= 0 && gy < S && gx >= 0 && gx < S && col < 130)
      v = sinf(img[((size_t)(b * 3 + ch) * S + gy) * S + gx]);
    simg[ch][row][col] = v;
  }
  __syncthreads();

  const int orow = tid >> 6;             // 0..3
  const int c2 = (tid & 63) * 2;         // 0..126
  const int p0 = orow * 128 + c2;        // local pixel index (strip-local)

  for (int c = 0; c < IN_C; ++c) {
    float w0 = wr[c * 3 + 0], w1 = wr[c * 3 + 1], w2 = wr[c * 3 + 2];
    float bc = bb[c];
    for (int i = tid; i < 792; i += 256) {
      int row = i / 132, col = i - row * 132;
      float v = w0 * simg[0][row][col] + w1 * simg[1][row][col] +
                w2 * simg[2][row][col] + bc;
      v = fminf(fmaxf(v, -256.f), 256.f);
      int gy = r0 - 1 + row, gx = col - 1;
      if (gy < 0 || gy >= S || gx < 0 || gx >= S) v = 0.f;  // exact zero pad
      xt[row][col] = v;
    }
    __syncthreads();   // xt ready
    float tvr[3] = {tv[c * 3], tv[c * 3 + 1], tv[c * 3 + 2]};
    float thr[3] = {th[c * 3], th[c * 3 + 1], th[c * 3 + 2]};
    // x values for this thread's two pixels
    unsigned short xv0 = f2bf(xt[orow + 1][c2 + 1]);
    unsigned short xv1 = f2bf(xt[orow + 1][c2 + 2]);
    // separable 3x3 depthwise
    float y0 = 0.f, y1 = 0.f;
    #pragma unroll
    for (int fi = 0; fi < 3; ++fi) {
      #pragma unroll
      for (int fj = 0; fj < 3; ++fj) {
        float t = tvr[fi] * thr[fj];
        y0 += t * xt[orow + fi][c2 + fj];
        y1 += t * xt[orow + fi][c2 + 1 + fj];
      }
    }
    int cl = c & 15;
    xstage[p0][cl] = xv0;  xstage[p0 + 1][cl] = xv1;
    ystage[p0][cl] = f2bf(y0);  ystage[p0 + 1][cl] = f2bf(y1);
    if ((c & 15) == 15) {
      __syncthreads();     // chunk complete
      int c0 = c & ~15;
      for (int idx = tid; idx < 2048; idx += 256) {
        int buf = idx >> 10;          // 0 = x, 1 = y
        int rem = idx & 1023;
        int px = rem >> 1, sub = rem & 1;
        int gy = r0 + (px >> 7), gx = px & 127;
        size_t g = (((size_t)(b * S + gy)) * S + gx) * 256 + c0 + sub * 8;
        const unsigned short* st = buf ? &ystage[px][sub * 8] : &xstage[px][sub * 8];
        uint4 v = *(const uint4*)st;
        *(uint4*)((buf ? yout : xout) + g) = v;
      }
      __syncthreads();     // stage reusable; also protects xt overwrite
    } else {
      __syncthreads();     // all xt reads done before next overwrite
    }
  }
}

// =====================================================================
// K4: fused K=512 WMMA GEMM  z = Wcat . [y_dw ; x]  + bias/lrelu/sqrt2/clamp
//     + stride-2 4x4 FIR downsample, all from LDS. 8 waves / workgroup.
// Channel-last activations: each staged B-row is a contiguous 64B chunk
// in global -> staging is ~5 b128 loads + 5 ds_store_b128 per thread.
// =====================================================================
#define NT    21
#define RPX   324
#define PXPAD 336

__global__ __launch_bounds__(256) void point_fir_kernel(
    const unsigned short* __restrict__ Wcat,  // [512,512] bf16
    const unsigned short* __restrict__ xin,   // [16,128,128,256] bf16 ch-last
    const unsigned short* __restrict__ yin,   // [16,128,128,256] bf16 ch-last
    const float* __restrict__ point_b,        // [512]
    const float* __restrict__ fir,            // [4], already /8
    float* __restrict__ out)                  // [16,512,64,64]
{
  __shared__ int poff[PXPAD];
  __shared__ __align__(16) unsigned short Bst[PXPAD * 32];  // [px][k] bf16
  __shared__ float zbuf[16 * RPX];
  __shared__ float biasl[128];

  const int tid = threadIdx.x;
  const int lane = tid & 31;
  const int wv = tid >> 5;
  const int bx = blockIdx.x;
  const int tile = bx >> 2;       // oc-block fastest -> L2 reuse of activations
  const int ocb = bx & 3;
  const int trow = tile >> 3, tcol = tile & 7;
  const int b = blockIdx.y;
  const int oc0 = ocb * 128;

  for (int i = tid; i < PXPAD; i += 256) {
    int off = -1;
    if (i < RPX) {
      int ry = i / 18, rx = i - ry * 18;
      int gy = trow * 16 - 1 + ry, gx = tcol * 16 - 1 + rx;
      if (gy >= 0 && gy < S && gx >= 0 && gx < S) off = gy * S + gx;
    }
    poff[i] = off;
  }
  if (tid < 128) biasl[tid] = point_b[oc0 + tid];
  float f0 = fir[0], f1 = fir[1], f2 = fir[2], f3 = fir[3];

  v8f acc[NT];
  const v8f vzero = {0.f, 0.f, 0.f, 0.f, 0.f, 0.f, 0.f, 0.f};
  #pragma unroll
  for (int t = 0; t < NT; ++t) acc[t] = vzero;

  // Per-lane operand addressing (16-bit WMMA layouts, ISA 7.12.2)
  const unsigned short* wrow = Wcat + (size_t)(oc0 + wv * 16 + (lane & 15)) * 512;
  const int aoff = (lane < 16) ? 0 : 8;    // A: K[0..7],[16..23] / K[8..15],[24..31]
  const int bklo = (lane < 16) ? 0 : 8;    // B: K[0..7],[16..23] / K[8..15],[24..31]
  const int bpx = lane & 15;
  const size_t pixbase = (size_t)b * S * S * 256;  // batch offset (elements)

  for (int ks = 0; ks < 16; ++ks) {
    const int kk0 = ks * 32;
    __syncthreads();             // Bst free (iter 0: poff ready)
    const unsigned short* srcp = (ks < 8) ? yin : xin;
    const int chb = (ks < 8) ? kk0 : kk0 - 256;
    // stage B tile: 336 px x 64B contiguous chunks (channel-last!)
    for (int idx = tid; idx < PXPAD * 4; idx += 256) {
      int px = idx >> 2, sub = idx & 3;
      int o = poff[px];
      uint4 v = {0u, 0u, 0u, 0u};
      if (o >= 0)
        v = *(const uint4*)(srcp + pixbase + (size_t)o * 256 + chb + sub * 8);
      *(uint4*)&Bst[px * 32 + sub * 8] = v;
    }
    __syncthreads();
    union { uint4 u[2]; v16bf v; } A;
    A.u[0] = *(const uint4*)(wrow + kk0 + aoff);
    A.u[1] = *(const uint4*)(wrow + kk0 + 16 + aoff);
    #pragma unroll
    for (int nt = 0; nt < NT; ++nt) {
      union { uint4 u[2]; v16bf v; } B;
      const unsigned short* bp = &Bst[(nt * 16 + bpx) * 32 + bklo];
      B.u[0] = *(const uint4*)(bp);
      B.u[1] = *(const uint4*)(bp + 16);
      acc[nt] = __builtin_amdgcn_wmma_f32_16x16x32_bf16(
          false, A.v, false, B.v, (short)0, acc[nt], false, false);
    }
  }

  // ---- epilogue: 8 phases of (16 out-ch -> LDS z, then FIR+store) ----
  const int orow0 = trow * 8, ocol0 = tcol * 8;
  for (int ph = 0; ph < 8; ++ph) {
    __syncthreads();
    if (wv == ph) {
      const int madd = (lane >> 4) << 3;   // D layout: VGPR r -> M=r / r+8
      #pragma unroll
      for (int nt = 0; nt < NT; ++nt) {
        int px = nt * 16 + bpx;
        if (px < RPX) {
          bool pad = (poff[px] < 0);
          #pragma unroll
          for (int r = 0; r < 8; ++r) {
            int m = r + madd;
            float v = acc[nt][r] + biasl[wv * 16 + m];
            v = (v >= 0.f) ? v : 0.2f * v;       // lrelu
            v *= 1.41421356237f;                 // gain sqrt(2)
            v = fminf(fmaxf(v, -256.f), 256.f);  // clamp
            if (pad) v = 0.f;                    // exact FIR zero padding
            zbuf[m * RPX + px] = v;
          }
        }
      }
    }
    __syncthreads();
    for (int i = tid; i < 1024; i += 256) {
      int ch = i >> 6, op = i & 63;
      int oy = op >> 3, ox = op & 7;
      const float* zr = &zbuf[ch * RPX + (2 * oy) * 18 + 2 * ox];
      float fw[4] = {f0, f1, f2, f3};
      float s = 0.f;
      #pragma unroll
      for (int fi = 0; fi < 4; ++fi) {
        float rs = fw[0] * zr[fi * 18 + 0] + fw[1] * zr[fi * 18 + 1] +
                   fw[2] * zr[fi * 18 + 2] + fw[3] * zr[fi * 18 + 3];
        s += fw[fi] * rs;
      }
      int oc = oc0 + ph * 16 + ch;
      out[(((size_t)b * OUT_C + oc) * SOUT + (orow0 + oy)) * SOUT + (ocol0 + ox)] = s;
    }
  }
}

// =====================================================================
extern "C" void kernel_launch(void* const* d_in, const int* in_sizes, int n_in,
                              void* d_out, int out_size, void* d_ws, size_t ws_size,
                              hipStream_t stream) {
  (void)in_sizes; (void)n_in; (void)out_size; (void)ws_size;
  const float* img        = (const float*)d_in[0];
  const float* fromrgb_w  = (const float*)d_in[1];
  const float* fromrgb_b  = (const float*)d_in[2];
  const float* freqs      = (const float*)d_in[3];
  const float* phases     = (const float*)d_in[4];
  const float* hz_outdim  = (const float*)d_in[5];
  const float* vt_outdim  = (const float*)d_in[6];
  const float* gauss_sig  = (const float*)d_in[7];
  const float* lr_weight0 = (const float*)d_in[8];
  const float* point_w    = (const float*)d_in[9];
  const float* point_b    = (const float*)d_in[10];
  const float* low_filter = (const float*)d_in[11];
  const float* gauss_x    = (const float*)d_in[12];
  const float* fir        = (const float*)d_in[13];

  char* ws = (char*)d_ws;
  float* taps_h = (float*)(ws + 0);
  float* taps_v = (float*)(ws + 4096);
  unsigned short* Wcat = (unsigned short*)(ws + 8192);
  const size_t XBYTES = (size_t)IMG_B * IN_C * S * S * 2;  // 134 MB bf16
  unsigned short* xbuf = (unsigned short*)(ws + (1u << 20));
  unsigned short* ybuf = (unsigned short*)(ws + (1u << 20) + XBYTES);

  gen_taps_kernel<<<1, 256, 0, stream>>>(freqs, phases, hz_outdim, vt_outdim,
                                         gauss_sig, low_filter, gauss_x,
                                         taps_h, taps_v);
  wcat_kernel<<<OUT_C, 256, 0, stream>>>(point_w, lr_weight0, Wcat);
  fromrgb_dw_kernel<<<dim3(S / 4, IMG_B), 256, 0, stream>>>(
      img, fromrgb_w, fromrgb_b, taps_h, taps_v, xbuf, ybuf);
  point_fir_kernel<<<dim3(256, IMG_B), 256, 0, stream>>>(
      Wcat, xbuf, ybuf, point_b, fir, (float*)d_out);
}
